// MTB_58179626992139
// MI455X (gfx1250) — matrix operand order repaired and checked by reading, not compiled, not run on hardware
//
#include <hip/hip_runtime.h>
#include <hip/hip_bf16.h>

typedef __attribute__((ext_vector_type(16))) _Float16 v16h;
typedef __attribute__((ext_vector_type(8)))  float    v8f;
typedef __attribute__((ext_vector_type(4)))  unsigned int u32x4;
typedef __attribute__((ext_vector_type(8)))  int      i32x8;
typedef __attribute__((ext_vector_type(4)))  int      i32x4;

#define Lpix 4096
#define CDIM 64
#define NCHUNK 64
#define CLEN   64

// ---------------------------------------------------------------------------
// TDM: DMA a [rows x cols] f32 tensor (row stride == cols) from global -> LDS.
// D# per ISA 8.3/8.4: group0 = {count, lds_addr, global_addr, type=2};
// group1 = {data_size=4B, tensor_dim0=cols, tensor_dim1=rows, tile=whole}.
// Issued by one wave; completion via TENSORcnt.
// ---------------------------------------------------------------------------
__device__ __forceinline__ void tdm_load_f32_rows(const float* __restrict__ gsrc,
                                                  void* lds_dst,
                                                  unsigned rows, unsigned cols) {
    unsigned long long ga = (unsigned long long)(size_t)gsrc;
    unsigned lds_addr = (unsigned)(size_t)lds_dst;       // low 32b = LDS offset
    u32x4 g0;
    g0[0] = 1u;                                          // count=1 (user D#)
    g0[1] = lds_addr;                                    // lds_addr [63:32]
    g0[2] = (unsigned)(ga & 0xFFFFFFFFu);                // global_addr lo
    g0[3] = (unsigned)((ga >> 32) & 0x01FFFFFFu)         // global_addr hi
          | (2u << 30);                                  // type=2 ("image")
    i32x8 g1;
    g1[0] = (int)(2u << 16);                             // data_size=2 -> 4 bytes
    g1[1] = (int)((cols & 0xFFFFu) << 16);               // tensor_dim0 lo16
    g1[2] = (int)(((cols >> 16) & 0xFFFFu) | ((rows & 0xFFFFu) << 16));
    g1[3] = (int)(((rows >> 16) & 0xFFFFu) | ((cols & 0xFFFFu) << 16)); // tile_dim0
    g1[4] = (int)(rows & 0xFFFFu);                       // tile_dim1 (tile_dim2=0)
    g1[5] = (int)cols;                                   // tensor_dim0_stride lo32
    g1[6] = 0;
    g1[7] = 0;
    i32x4 g2 = {0, 0, 0, 0};
    i32x4 g3 = {0, 0, 0, 0};
#if defined(__clang_major__) && (__clang_major__ >= 23)
    i32x8 g4 = {0, 0, 0, 0, 0, 0, 0, 0};
    __builtin_amdgcn_tensor_load_to_lds(g0, g1, g2, g3, g4, 0);
#else
    __builtin_amdgcn_tensor_load_to_lds(g0, g1, g2, g3, 0);
#endif
}

// ---------------------------------------------------------------------------
// WMMA fragment loaders, wave32; A and B both use contiguous k = 16*(lane/16) + s
// ---------------------------------------------------------------------------
__device__ __forceinline__ v16h load_a_frag(const float* __restrict__ row, int k0, int lane) {
    int kb = k0 + ((lane < 16) ? 0 : 16);
    v16h f;
#pragma unroll
    for (int i = 0; i < 8; ++i) {
        f[i]     = (_Float16)row[kb + i];
        f[i + 8] = (_Float16)row[kb + 8 + i];
    }
    return f;
}

__device__ __forceinline__ v16h load_b_frag_lds(const _Float16* __restrict__ sW,
                                                int rowB, int k0, int lane) {
    const _Float16* p = sW + rowB * 64 + k0 + ((lane < 16) ? 0 : 16);
    v16h f;
#pragma unroll
    for (int i = 0; i < 16; ++i) f[i] = p[i];
    return f;
}

// ---------------------------------------------------------------------------
// Y[M,N] = X[M,64] @ W[N,64]^T.  W pulled into LDS by the TDM, converted to
// f16 in place; each wave converts its A fragments once and sweeps N tiles.
// ---------------------------------------------------------------------------
__global__ void gemm_xwt_wmma(const float* __restrict__ X, const float* __restrict__ W,
                              float* __restrict__ Y, int M, int N) {
    __shared__ float sWf32[192 * 64];            // 48 KB; f16 view reuses storage
    _Float16* sW = (_Float16*)sWf32;
    int tid = threadIdx.x;
    if (tid == 0) {
        tdm_load_f32_rows(W, sWf32, (unsigned)N, 64u);
        __builtin_amdgcn_s_wait_tensorcnt(0);
    }
    __syncthreads();
    // in-place f32 -> f16 (register-buffered; barrier separates read & write)
    float tmp[48];                               // N*64/256 <= 48
    int cnt = 0;
    for (int i = tid; i < N * 64; i += 256) tmp[cnt++] = sWf32[i];
    __syncthreads();
    cnt = 0;
    for (int i = tid; i < N * 64; i += 256) sW[i] = (_Float16)tmp[cnt++];
    __syncthreads();

    int lane   = tid & 31;
    int wave   = blockIdx.x * (blockDim.x >> 5) + (tid >> 5);
    int nwaves = gridDim.x * (blockDim.x >> 5);
    int mt = M >> 4, nt = N >> 4;
    for (int tm = wave; tm < mt; tm += nwaves) {
        const float* arow = X + (size_t)(tm * 16 + (lane & 15)) * 64;
        v16h a0 = load_a_frag(arow, 0, lane);
        v16h a1 = load_a_frag(arow, 32, lane);
        int mbase = tm * 16 + ((lane < 16) ? 0 : 8);
        for (int tn = 0; tn < nt; ++tn) {
            int rowB = tn * 16 + (lane & 15);
            v16h b0 = load_b_frag_lds(sW, rowB, 0, lane);
            v16h b1 = load_b_frag_lds(sW, rowB, 32, lane);
            v8f acc = {};
            acc = __builtin_amdgcn_wmma_f32_16x16x32_f16(false, a0, false, b0,
                                                         (short)0, acc, false, false);
            acc = __builtin_amdgcn_wmma_f32_16x16x32_f16(false, a1, false, b1,
                                                         (short)0, acc, false, false);
            int nn = tn * 16 + (lane & 15);
#pragma unroll
            for (int r = 0; r < 8; ++r)
                Y[(size_t)(mbase + r) * N + nn] = acc[r];
        }
    }
}

// ---------------------------------------------------------------------------
// x_dbl: per (b,k): [4096,64] @ xproj[k][20,64]^T -> [4096,20]
// TDM stages all 4 direction weights (80x64 f32); converted to f16 padded
// to 32 rows so the WMMA sweep is branch-free.
// ---------------------------------------------------------------------------
__global__ void xdbl_gemm_wmma(const float* __restrict__ XS, const float* __restrict__ XPROJ,
                               float* __restrict__ XDBL) {
    __shared__ float    sXP[80 * 64];            // 20 KB (TDM destination)
    __shared__ _Float16 sW[4 * 32 * 64];         // 16 KB (padded f16)
    int tid = threadIdx.x;
    if (tid == 0) {
        tdm_load_f32_rows(XPROJ, sXP, 80u, 64u);
        __builtin_amdgcn_s_wait_tensorcnt(0);
    }
    __syncthreads();
    for (int i = tid; i < 4 * 32 * 64; i += blockDim.x) {
        int kk = i >> 11;
        int r  = (i >> 6) & 31;
        int c  = i & 63;
        sW[i] = (r < 20) ? (_Float16)sXP[((size_t)kk * 20 + r) * 64 + c] : (_Float16)0.0f;
    }
    __syncthreads();
    int lane   = tid & 31;
    int wave   = blockIdx.x * (blockDim.x >> 5) + (tid >> 5);
    int nwaves = gridDim.x * (blockDim.x >> 5);
    const int mtTotal = 16 * (Lpix / 16);
    for (int t = wave; t < mtTotal; t += nwaves) {
        int bk = t >> 8;
        int tm = t & 255;
        int k  = bk & 3;
        const float* arow = XS + ((size_t)bk * Lpix + tm * 16 + (lane & 15)) * CDIM;
        v16h a0 = load_a_frag(arow, 0, lane);
        v16h a1 = load_a_frag(arow, 32, lane);
        int mbase = tm * 16 + ((lane < 16) ? 0 : 8);
        const _Float16* wbase = sW + k * 32 * 64;
#pragma unroll
        for (int tn = 0; tn < 2; ++tn) {
            int rowB = tn * 16 + (lane & 15);
            v16h b0 = load_b_frag_lds(wbase, rowB, 0, lane);
            v16h b1 = load_b_frag_lds(wbase, rowB, 32, lane);
            v8f acc = {};
            acc = __builtin_amdgcn_wmma_f32_16x16x32_f16(false, a0, false, b0,
                                                         (short)0, acc, false, false);
            acc = __builtin_amdgcn_wmma_f32_16x16x32_f16(false, a1, false, b1,
                                                         (short)0, acc, false, false);
            int nn = tn * 16 + (lane & 15);
            if (nn < 20) {
#pragma unroll
                for (int r = 0; r < 8; ++r)
                    XDBL[((size_t)bk * Lpix + mbase + r) * 20 + nn] = acc[r];
            }
        }
    }
}

// ---------------------------------------------------------------------------
// LayerNorm over 64 channels; one pixel per 64-thread block.
// ---------------------------------------------------------------------------
__global__ void layernorm64(const float* __restrict__ in, const float* __restrict__ w,
                            const float* __restrict__ b, float* __restrict__ out,
                            int is_nchw) {
    __shared__ float sbuf[CDIM];
    __shared__ float smu, svar;
    int m = blockIdx.x;
    int d = threadIdx.x;
    int bn = m >> 12, pix = m & (Lpix - 1);
    float v = is_nchw ? in[((size_t)bn * CDIM + d) * Lpix + pix]
                      : in[(size_t)m * CDIM + d];
    sbuf[d] = v;
    __syncthreads();
    if (d == 0) {
        float s = 0.f;
        for (int i = 0; i < CDIM; ++i) s += sbuf[i];
        smu = s * (1.0f / CDIM);
    }
    __syncthreads();
    float diff = v - smu;
    sbuf[d] = diff * diff;
    __syncthreads();
    if (d == 0) {
        float s = 0.f;
        for (int i = 0; i < CDIM; ++i) s += sbuf[i];
        svar = s * (1.0f / CDIM);
    }
    __syncthreads();
    out[(size_t)m * CDIM + d] = diff * rsqrtf(svar + 1e-5f) * w[d] + b[d];
}

// ---------------------------------------------------------------------------
// Depthwise 3x3 SAME + bias + SiLU (input: cols 0:64 of NHWC [m,128]).
// ---------------------------------------------------------------------------
__global__ void conv_dw_silu(const float* __restrict__ XZ, const float* __restrict__ CW,
                             const float* __restrict__ CB, float* __restrict__ XC) {
    int idx = blockIdx.x * blockDim.x + threadIdx.x;
    if (idx >= 4 * Lpix * CDIM) return;
    int d = idx & 63, m = idx >> 6;
    int bn = m >> 12, pix = m & (Lpix - 1);
    int h_ = pix >> 6, w_ = pix & 63;
    float acc = CB[d];
#pragma unroll
    for (int dy = -1; dy <= 1; ++dy)
#pragma unroll
        for (int dx = -1; dx <= 1; ++dx) {
            int hh = h_ + dy, ww = w_ + dx;
            if (hh >= 0 && hh < 64 && ww >= 0 && ww < 64) {
                float wv = CW[(d * 3 + (dy + 1)) * 3 + (dx + 1)];
                acc += wv * XZ[((size_t)bn * Lpix + hh * 64 + ww) * 128 + d];
            }
        }
    XC[idx] = acc / (1.0f + __expf(-acc));
}

// ---------------------------------------------------------------------------
// Build 4-direction sequences, l-major: XS[b,k,l,d].
// ---------------------------------------------------------------------------
__global__ void build_xs(const float* __restrict__ XC, float* __restrict__ XS) {
    int idx = blockIdx.x * blockDim.x + threadIdx.x;
    if (idx >= 16 * Lpix * CDIM) return;
    int d = idx & 63;
    int rest = idx >> 6;
    int l = rest & (Lpix - 1);
    int bk = rest >> 12;
    int b = bk >> 2, k = bk & 3;
    int le = (k >= 2) ? (Lpix - 1 - l) : l;
    int h_, w_;
    if ((k & 1) == 0) { h_ = le >> 6; w_ = le & 63; }
    else              { w_ = le >> 6; h_ = le & 63; }
    XS[idx] = XC[((size_t)b * Lpix + h_ * 64 + w_) * CDIM + d];
}

// ---------------------------------------------------------------------------
// Chunked linear-recurrence scan: h[l] = dA[l]*h[l-1] + du[l]*B[l]
// ---------------------------------------------------------------------------
__global__ void scan_pass1(const float* __restrict__ XS, const float* __restrict__ XDBL,
                           const float* __restrict__ DTW, const float* __restrict__ DTB,
                           const float* __restrict__ ALOG,
                           float* __restrict__ DTBUF,
                           float* __restrict__ PB, float* __restrict__ SB) {
    int bkc = blockIdx.x;
    int bk = bkc >> 6;
    int ch = bkc & 63;
    int k  = bk & 3;
    int d  = threadIdx.x;
    const float* dw = DTW + ((size_t)k * CDIM + d) * 4;
    float w0 = dw[0], w1 = dw[1], w2 = dw[2], w3 = dw[3];
    float dtb = DTB[k * CDIM + d];
    float a[8], P[8], S[8];
#pragma unroll
    for (int n = 0; n < 8; ++n) {
        a[n] = -__expf(ALOG[((size_t)k * CDIM + d) * 8 + n]);
        P[n] = 1.f;
        S[n] = 0.f;
    }
    int l0 = ch * CLEN;
    const float* xsrow = XS + ((size_t)bk * Lpix + l0) * CDIM;
    const float* xdrow = XDBL + ((size_t)bk * Lpix + l0) * 20;
    float* dtrow = DTBUF + ((size_t)bk * Lpix + l0) * CDIM;
    for (int i = 0; i < CLEN; ++i) {
        const float* xd = xdrow + (size_t)i * 20;
        float u  = xsrow[(size_t)i * CDIM + d];
        float dt = xd[0] * w0 + xd[1] * w1 + xd[2] * w2 + xd[3] * w3 + dtb;
        dt = (dt > 20.f) ? dt : log1pf(__expf(dt));
        dtrow[(size_t)i * CDIM + d] = dt;
        float du = dt * u;
#pragma unroll
        for (int n = 0; n < 8; ++n) {
            float dA = __expf(dt * a[n]);
            S[n] = dA * S[n] + du * xd[4 + n];
            P[n] *= dA;
        }
        if ((i & 7) == 0 && (i + 8) < CLEN) {
            __builtin_prefetch(xsrow + (size_t)(i + 8) * CDIM + d, 0, 1);
            __builtin_prefetch(xdrow + (size_t)(i + 8) * 20, 0, 1);
        }
    }
    size_t base = (((size_t)bk * CDIM + d) * 8) * NCHUNK + ch;
#pragma unroll
    for (int n = 0; n < 8; ++n) {
        PB[base + (size_t)n * NCHUNK] = P[n];
        SB[base + (size_t)n * NCHUNK] = S[n];
    }
}

__global__ void scan_pass2(const float* __restrict__ PB, const float* __restrict__ SB,
                           float* __restrict__ HI) {
    int idx = blockIdx.x * blockDim.x + threadIdx.x;
    if (idx >= 16 * CDIM * 8) return;
    size_t base = (size_t)idx * NCHUNK;
    float h = 0.f;
    for (int c = 0; c < NCHUNK; ++c) {
        HI[base + c] = h;
        h = SB[base + c] + PB[base + c] * h;
    }
}

__global__ void scan_pass3(const float* __restrict__ XS, const float* __restrict__ XDBL,
                           const float* __restrict__ DTBUF, const float* __restrict__ ALOG,
                           const float* __restrict__ DP, const float* __restrict__ HI,
                           float* __restrict__ Y) {
    int bkc = blockIdx.x;
    int bk = bkc >> 6;
    int ch = bkc & 63;
    int k  = bk & 3;
    int d  = threadIdx.x;
    float Dv = DP[k * CDIM + d];
    float a[8], hst[8];
    size_t hbase = (((size_t)bk * CDIM + d) * 8) * NCHUNK + ch;
#pragma unroll
    for (int n = 0; n < 8; ++n) {
        a[n]   = -__expf(ALOG[((size_t)k * CDIM + d) * 8 + n]);
        hst[n] = HI[hbase + (size_t)n * NCHUNK];
    }
    int l0 = ch * CLEN;
    const float* xsrow = XS + ((size_t)bk * Lpix + l0) * CDIM;
    const float* xdrow = XDBL + ((size_t)bk * Lpix + l0) * 20;
    const float* dtrow = DTBUF + ((size_t)bk * Lpix + l0) * CDIM;
    float* yrow = Y + ((size_t)bk * Lpix + l0) * CDIM;
    for (int i = 0; i < CLEN; ++i) {
        const float* xd = xdrow + (size_t)i * 20;
        float u  = xsrow[(size_t)i * CDIM + d];
        float dt = dtrow[(size_t)i * CDIM + d];
        float du = dt * u;
        float y = 0.f;
#pragma unroll
        for (int n = 0; n < 8; ++n) {
            float dA = __expf(dt * a[n]);
            hst[n] = dA * hst[n] + du * xd[4 + n];
            y += hst[n] * xd[12 + n];
        }
        yrow[(size_t)i * CDIM + d] = y + Dv * u;
        if ((i & 7) == 0 && (i + 8) < CLEN) {
            __builtin_prefetch(xsrow + (size_t)(i + 8) * CDIM + d, 0, 1);
            __builtin_prefetch(xdrow + (size_t)(i + 8) * 20, 0, 1);
        }
    }
}

// ---------------------------------------------------------------------------
// Combine 4 directions + LN(onw,onb) + silu(z) gate.
// ---------------------------------------------------------------------------
__global__ void combine_ln_gate(const float* __restrict__ YD, const float* __restrict__ XZ,
                                const float* __restrict__ onw, const float* __restrict__ onb,
                                float* __restrict__ YG) {
    __shared__ float sbuf[CDIM];
    __shared__ float smu, svar;
    int m = blockIdx.x;
    int d = threadIdx.x;
    int bn = m >> 12, pix = m & (Lpix - 1);
    int h_ = pix >> 6, w_ = pix & 63;
    int l_hw = pix, l_wh = w_ * 64 + h_;
    float v = YD[(((size_t)bn * 4 + 0) * Lpix + l_hw) * CDIM + d]
            + YD[(((size_t)bn * 4 + 2) * Lpix + (Lpix - 1 - l_hw)) * CDIM + d]
            + YD[(((size_t)bn * 4 + 1) * Lpix + l_wh) * CDIM + d]
            + YD[(((size_t)bn * 4 + 3) * Lpix + (Lpix - 1 - l_wh)) * CDIM + d];
    sbuf[d] = v;
    __syncthreads();
    if (d == 0) {
        float s = 0.f;
        for (int i = 0; i < CDIM; ++i) s += sbuf[i];
        smu = s * (1.0f / CDIM);
    }
    __syncthreads();
    float diff = v - smu;
    sbuf[d] = diff * diff;
    __syncthreads();
    if (d == 0) {
        float s = 0.f;
        for (int i = 0; i < CDIM; ++i) s += sbuf[i];
        svar = s * (1.0f / CDIM);
    }
    __syncthreads();
    float yn = diff * rsqrtf(svar + 1e-5f) * onw[d] + onb[d];
    float z  = XZ[(size_t)m * 128 + 64 + d];
    YG[(size_t)m * CDIM + d] = yn * (z / (1.0f + __expf(-z)));
}

__global__ void add_skip_nchw(const float* __restrict__ T, const float* __restrict__ SKIP,
                              float* __restrict__ OUT) {
    int idx = blockIdx.x * blockDim.x + threadIdx.x;
    if (idx >= 4 * CDIM * Lpix) return;
    int pix = idx & (Lpix - 1);
    int c   = (idx >> 12) & 63;
    int bn  = idx >> 18;
    OUT[idx] = T[((size_t)bn * Lpix + pix) * CDIM + c] + SKIP[idx];
}

// ---------------------------------------------------------------------------
// Channel attention stats + softmax (LDS ds_add_f32 reduction).
// ---------------------------------------------------------------------------
__global__ void attn_stats(const float* __restrict__ QKV, const float* __restrict__ TEMP,
                           float* __restrict__ ATTN) {
    __shared__ float red[80];
    int bh = blockIdx.x;
    int bn = bh >> 3, hd = bh & 7;
    int tid = threadIdx.x;
    float qq[8], kk[8], qk[64];
#pragma unroll
    for (int i = 0; i < 8; ++i) { qq[i] = 0.f; kk[i] = 0.f; }
#pragma unroll
    for (int i = 0; i < 64; ++i) qk[i] = 0.f;
    const float* base = QKV + (size_t)bn * Lpix * 192 + hd * 8;
    for (int l = tid; l < Lpix; l += 256) {
        const float* row = base + (size_t)l * 192;
        float q[8], kv[8];
#pragma unroll
        for (int c = 0; c < 8; ++c) { q[c] = row[c]; kv[c] = row[64 + c]; }
#pragma unroll
        for (int c = 0; c < 8; ++c) {
            qq[c] += q[c] * q[c];
            kk[c] += kv[c] * kv[c];
#pragma unroll
            for (int dd = 0; dd < 8; ++dd) qk[c * 8 + dd] += q[c] * kv[dd];
        }
    }
    if (tid < 80) red[tid] = 0.f;
    __syncthreads();
#pragma unroll
    for (int i = 0; i < 8; ++i) {
        atomicAdd(&red[i], qq[i]);
        atomicAdd(&red[8 + i], kk[i]);
    }
#pragma unroll
    for (int i = 0; i < 64; ++i) atomicAdd(&red[16 + i], qk[i]);
    __syncthreads();
    if (tid == 0) {
        float t = TEMP[hd];
        float qn[8], kn[8];
        for (int c = 0; c < 8; ++c) {
            qn[c] = fmaxf(sqrtf(red[c]), 1e-12f);
            kn[c] = fmaxf(sqrtf(red[8 + c]), 1e-12f);
        }
        for (int c = 0; c < 8; ++c) {
            float logit[8], mx = -3.4e38f;
            for (int dd = 0; dd < 8; ++dd) {
                logit[dd] = red[16 + c * 8 + dd] / (qn[c] * kn[dd]) * t;
                mx = fmaxf(mx, logit[dd]);
            }
            float s = 0.f;
            for (int dd = 0; dd < 8; ++dd) { logit[dd] = __expf(logit[dd] - mx); s += logit[dd]; }
            float inv = 1.0f / s;
            for (int dd = 0; dd < 8; ++dd)
                ATTN[(size_t)bh * 64 + c * 8 + dd] = logit[dd] * inv;
        }
    }
}

__global__ void attn_apply(const float* __restrict__ QKV, const float* __restrict__ ATTN,
                           float* __restrict__ CAOUT) {
    int idx = blockIdx.x * blockDim.x + threadIdx.x;
    if (idx >= 4 * Lpix * CDIM) return;
    int o = idx & 63, m = idx >> 6;
    int bn = m >> 12, hd = o >> 3, ch = o & 7;
    const float* at = ATTN + (((size_t)bn * 8 + hd) * 8 + ch) * 8;
    const float* vr = QKV + (size_t)m * 192 + 128 + hd * 8;
    float s = 0.f;
#pragma unroll
    for (int dd = 0; dd < 8; ++dd) s += at[dd] * vr[dd];
    CAOUT[idx] = s;
}

__global__ void final_add(const float* __restrict__ PROJ, const float* __restrict__ SKIP,
                          float* __restrict__ OUT) {
    int idx = blockIdx.x * blockDim.x + threadIdx.x;
    if (idx >= 4 * CDIM * Lpix) return;
    int pix = idx & (Lpix - 1);
    int c   = (idx >> 12) & 63;
    int bn  = idx >> 18;
    OUT[idx] = PROJ[((size_t)bn * Lpix + pix) * CDIM + c] + SKIP[idx];
}

// ---------------------------------------------------------------------------
extern "C" void kernel_launch(void* const* d_in, const int* in_sizes, int n_in,
                              void* d_out, int out_size, void* d_ws, size_t ws_size,
                              hipStream_t stream) {
    (void)in_sizes; (void)n_in; (void)out_size; (void)ws_size;
    const float* x      = (const float*)d_in[0];
    const float* ln1w   = (const float*)d_in[1];
    const float* ln1b   = (const float*)d_in[2];
    const float* ln2w   = (const float*)d_in[3];
    const float* ln2b   = (const float*)d_in[4];
    const float* in_w   = (const float*)d_in[5];
    const float* conv_w = (const float*)d_in[6];
    const float* conv_b = (const float*)d_in[7];
    const float* xprojw = (const float*)d_in[8];
    const float* dt_w   = (const float*)d_in[9];
    const float* dt_b   = (const float*)d_in[10];
    const float* Alog   = (const float*)d_in[11];
    const float* Dp     = (const float*)d_in[12];
    const float* onw    = (const float*)d_in[13];
    const float* onb    = (const float*)d_in[14];
    const float* outw   = (const float*)d_in[15];
    const float* temp   = (const float*)d_in[16];
    const float* qkv_w  = (const float*)d_in[17];
    const float* proj_w = (const float*)d_in[18];
    float* out = (float*)d_out;

    const int M = 4 * Lpix;
    float* ws = (float*)d_ws;
    float* t_a    = ws;                 // 1,048,576
    float* t_b    = ws + 1048576;       // 1,048,576
    float* xz     = ws + 2097152;       // 2,097,152
    float* xconv  = ws + 4194304;       // 1,048,576
    float* xs     = ws + 5242880;       // 4,194,304
    float* xdbl   = ws + 9437184;       // 1,310,720
    float* ydir   = ws + 10747904;      // 4,194,304
    float* yg     = ws + 14942208;      // 1,048,576
    float* outnc  = ws + 15990784;      // 1,048,576
    float* attn   = ws + 17039360;      // 2,048
    float* dtbuf  = ws + 17041408;      // 4,194,304
    float* Pbuf   = ws + 21235712;      // 524,288
    float* Sbuf   = ws + 21760000;      // 524,288
    float* Hinit  = ws + 22284288;      // 524,288
    float* qkv    = xs;                 // alias
    float* caout  = ydir;               // alias
    float* proj   = xz;                 // alias

    const int EW = 256;

    layernorm64<<<M, 64, 0, stream>>>(x, ln1w, ln1b, t_a, 1);

    float* tin = t_a;
    float* tout = t_b;
    for (int layer = 0; layer < 2; ++layer) {
        gemm_xwt_wmma<<<128, EW, 0, stream>>>(
            tin, in_w + (size_t)layer * 128 * 64, xz, M, 128);
        conv_dw_silu<<<(M * CDIM + EW - 1) / EW, EW, 0, stream>>>(
            xz, conv_w + (size_t)layer * 64 * 9, conv_b + (size_t)layer * 64, xconv);
        build_xs<<<(16 * Lpix * CDIM + EW - 1) / EW, EW, 0, stream>>>(xconv, xs);
        xdbl_gemm_wmma<<<512, EW, 0, stream>>>(
            xs, xprojw + (size_t)layer * 4 * 20 * 64, xdbl);
        scan_pass1<<<16 * NCHUNK, 64, 0, stream>>>(
            xs, xdbl,
            dt_w + (size_t)layer * 4 * 64 * 4, dt_b + (size_t)layer * 4 * 64,
            Alog + (size_t)layer * 256 * 8, dtbuf, Pbuf, Sbuf);
        scan_pass2<<<(16 * CDIM * 8 + EW - 1) / EW, EW, 0, stream>>>(Pbuf, Sbuf, Hinit);
        scan_pass3<<<16 * NCHUNK, 64, 0, stream>>>(
            xs, xdbl, dtbuf, Alog + (size_t)layer * 256 * 8,
            Dp + (size_t)layer * 256, Hinit, ydir);
        combine_ln_gate<<<M, 64, 0, stream>>>(
            ydir, xz, onw + (size_t)layer * 64, onb + (size_t)layer * 64, yg);
        gemm_xwt_wmma<<<128, EW, 0, stream>>>(
            yg, outw + (size_t)layer * 64 * 64, tout, M, 64);
        float* tmp = tin; tin = tout; tout = tmp;
    }
    add_skip_nchw<<<(4 * CDIM * Lpix + EW - 1) / EW, EW, 0, stream>>>(tin, x, outnc);
    layernorm64<<<M, 64, 0, stream>>>(outnc, ln2w, ln2b, tout, 1);
    gemm_xwt_wmma<<<128, EW, 0, stream>>>(tout, qkv_w, qkv, M, 192);
    attn_stats<<<32, 256, 0, stream>>>(qkv, temp, attn);
    attn_apply<<<(M * CDIM + EW - 1) / EW, EW, 0, stream>>>(qkv, attn, caout);
    gemm_xwt_wmma<<<128, EW, 0, stream>>>(caout, proj_w, proj, M, 64);
    final_add<<<(4 * CDIM * Lpix + EW - 1) / EW, EW, 0, stream>>>(proj, outnc, out);
}
